// HyperRHN_70188355551602
// MI455X (gfx1250) — matrix operand order repaired and checked by reading, not compiled, not run on hardware
//
#include <hip/hip_runtime.h>
#include <hip/hip_bf16.h>

// ---------------------------------------------------------------------------
// HyperRHN forward for MI455X (gfx1250): f16 WMMA GEMMs with fused highway
// gates. Weights converted once to f16 (L2-resident: all weights ~90MB f16 fit
// in the 192MB L2). GEMM waves block M=32 so every B fragment feeds 2 WMMAs,
// and each k-step is structured as load-burst -> WMMA-burst for pipelining.
// ---------------------------------------------------------------------------

typedef __attribute__((ext_vector_type(16))) _Float16 v16h;
typedef __attribute__((ext_vector_type(8)))  _Float16 v8h;
typedef __attribute__((ext_vector_type(4)))  _Float16 v4h;
typedef __attribute__((ext_vector_type(8)))  float    v8f;

#define RHN_DEPTH 4
#define RHN_HH    512
#define RHN_HN    2048
#define RHN_ED    1024
#define RHN_B     512

#define RHN_WMMA(a, b, c) \
  __builtin_amdgcn_wmma_f32_16x16x32_f16(false, (a), false, (b), (short)0, (c), false, false)

// ---- WMMA fragment loaders (layouts per CDNA5 ISA 7.12.2, wave32) ----------
// A matrix 16x32 f16, row-major source [ld = K]:
//   lane = half*16 + m16 ; halves 0..7  -> K = k + half*8 + h
//                          halves 8..15 -> K = k + 16 + half*8 + (h-8)
__device__ __forceinline__ v16h rhn_load_frag_a(const _Float16* __restrict__ Arow,
                                                int ld, int k, int half, int m16) {
  const _Float16* p = Arow + (size_t)m16 * (size_t)ld + (size_t)(k + half * 8);
  v8h lo = *(const v8h*)(p);
  v8h hi = *(const v8h*)(p + 16);
  return __builtin_shufflevector(lo, hi, 0, 1, 2, 3, 4, 5, 6, 7,
                                 8, 9, 10, 11, 12, 13, 14, 15);
}

// B matrix 32x16 f16 where B[k][n] = W[n][k], W row-major [ld = K]:
//   lane = half*16 + m16 (column n = m16); halves 0..7  -> K = k + half*8 + h
//                                          halves 8..15 -> K = k + 16 + half*8 + (h-8)
__device__ __forceinline__ v16h rhn_load_frag_b(const _Float16* __restrict__ Wtile,
                                                int ld, int k, int half, int m16) {
  const _Float16* p = Wtile + (size_t)m16 * (size_t)ld + (size_t)(k + half * 8);
  v8h lo = *(const v8h*)(p);
  v8h hi = *(const v8h*)(p + 16);
  return __builtin_shufflevector(lo, hi, 0, 1, 2, 3, 4, 5, 6, 7,
                                 8, 9, 10, 11, 12, 13, 14, 15);
}

// ---- fused highway GEMM + gate --------------------------------------------
// Wave tile: 32 rows (2 M-frags) x 32 cols, plus the paired tt columns
// (col + N). 8 accumulators; per k-step: 6 fragment loads -> 8 WMMAs.
//   pre   = A1@W1.T [+ A2@W2.T]
//   hh    = tanh((pre[:,n]+b)*z?)  tt = sigmoid((pre[:,N+n]+b)*z?)
//   out   = hh*tt + s_in*(1-tt)
__global__ __launch_bounds__(32)
void hyperrhn_highway_wmma(const _Float16* __restrict__ A1,
                           const _Float16* __restrict__ W1,
                           const float*    __restrict__ b1, int K1,
                           const _Float16* __restrict__ A2,
                           const _Float16* __restrict__ W2,
                           const float*    __restrict__ b2, int K2,
                           const float*    __restrict__ zscale,
                           const float*    __restrict__ s_in,
                           float*          __restrict__ s_out,
                           _Float16*       __restrict__ s_out16,
                           int N) {
  const int lane = threadIdx.x;
  const int half = lane >> 4;
  const int m16  = lane & 15;
  const int m0   = blockIdx.x * 32;  // 2 M tiles of 16
  const int n0   = blockIdx.y * 32;  // 2 N tiles of 16 (+ paired tt tiles)

  v8f cHH[2][2], cTT[2][2];  // [mi][nj]
#pragma unroll
  for (int mi = 0; mi < 2; ++mi)
#pragma unroll
    for (int nj = 0; nj < 2; ++nj)
#pragma unroll
      for (int e = 0; e < 8; ++e) { cHH[mi][nj][e] = 0.0f; cTT[mi][nj][e] = 0.0f; }

#pragma unroll
  for (int phase = 0; phase < 2; ++phase) {
    const _Float16* A = phase ? A2 : A1;
    const _Float16* W = phase ? W2 : W1;
    const int       K = phase ? K2 : K1;
    if (A == nullptr) continue;  // uniform branch: EXEC stays all-ones
    const _Float16* Arow0 = A + (size_t)m0 * (size_t)K;
    const _Float16* Arow1 = A + (size_t)(m0 + 16) * (size_t)K;
    const _Float16* Wh0 = W + (size_t)(n0) * (size_t)K;
    const _Float16* Wh1 = W + (size_t)(n0 + 16) * (size_t)K;
    const _Float16* Wt0 = W + (size_t)(N + n0) * (size_t)K;
    const _Float16* Wt1 = W + (size_t)(N + n0 + 16) * (size_t)K;
    for (int k = 0; k < K; k += 32) {
      // ---- load burst (12 x b128, compiler can clause these) ----
      const v16h a0 = rhn_load_frag_a(Arow0, K, k, half, m16);
      const v16h a1 = rhn_load_frag_a(Arow1, K, k, half, m16);
      const v16h b0 = rhn_load_frag_b(Wh0, K, k, half, m16);
      const v16h b1v = rhn_load_frag_b(Wh1, K, k, half, m16);
      const v16h t0 = rhn_load_frag_b(Wt0, K, k, half, m16);
      const v16h t1 = rhn_load_frag_b(Wt1, K, k, half, m16);
      // ---- WMMA burst: every fragment used twice ----
      cHH[0][0] = RHN_WMMA(a0, b0, cHH[0][0]);
      cHH[1][0] = RHN_WMMA(a1, b0, cHH[1][0]);
      cHH[0][1] = RHN_WMMA(a0, b1v, cHH[0][1]);
      cHH[1][1] = RHN_WMMA(a1, b1v, cHH[1][1]);
      cTT[0][0] = RHN_WMMA(a0, t0, cTT[0][0]);
      cTT[1][0] = RHN_WMMA(a1, t0, cTT[1][0]);
      cTT[0][1] = RHN_WMMA(a0, t1, cTT[0][1]);
      cTT[1][1] = RHN_WMMA(a1, t1, cTT[1][1]);
    }
  }

  // Epilogue: C/D element (VGPR r, lane) = (M = r + 8*half, Ncol = m16)
#pragma unroll
  for (int nj = 0; nj < 2; ++nj) {
    const int col = n0 + 16 * nj + m16;
    float bhh = b1[col];
    float btt = b1[N + col];
    if (b2 != nullptr) { bhh += b2[col]; btt += b2[N + col]; }
#pragma unroll
    for (int mi = 0; mi < 2; ++mi) {
#pragma unroll
      for (int r = 0; r < 8; ++r) {
        const int    row = m0 + 16 * mi + r + 8 * half;
        const size_t idx = (size_t)row * (size_t)N + (size_t)col;
        float hp = cHH[mi][nj][r] + bhh;
        float tp = cTT[mi][nj][r] + btt;
        if (zscale != nullptr) {
          const float zv = zscale[idx];
          hp *= zv; tp *= zv;
        }
        const float hh = tanhf(hp);
        const float tt = 1.0f / (1.0f + expf(-tp));
        const float s  = s_in[idx];
        const float o  = hh * tt + s * (1.0f - tt);
        s_out[idx]   = o;
        s_out16[idx] = (_Float16)o;
      }
    }
  }
}

// ---- plain GEMM + bias (z = sHyper @ Wup.T + bup) -------------------------
// Wave tile: 32 rows x 64 cols; 8 accumulators; per k-step 6 loads -> 8 WMMAs.
__global__ __launch_bounds__(32)
void hyperrhn_gemm_bias_wmma(const _Float16* __restrict__ A,
                             const _Float16* __restrict__ W,
                             const float*    __restrict__ bias,
                             float*          __restrict__ out,
                             int N, int K) {
  const int lane = threadIdx.x;
  const int half = lane >> 4;
  const int m16  = lane & 15;
  const int m0   = blockIdx.x * 32;
  const int n0   = blockIdx.y * 64;

  v8f c[2][4];
#pragma unroll
  for (int mi = 0; mi < 2; ++mi)
#pragma unroll
    for (int nj = 0; nj < 4; ++nj)
#pragma unroll
      for (int e = 0; e < 8; ++e) c[mi][nj][e] = 0.0f;

  const _Float16* Arow0 = A + (size_t)m0 * (size_t)K;
  const _Float16* Arow1 = A + (size_t)(m0 + 16) * (size_t)K;
  for (int k = 0; k < K; k += 32) {
    const v16h a0 = rhn_load_frag_a(Arow0, K, k, half, m16);
    const v16h a1 = rhn_load_frag_a(Arow1, K, k, half, m16);
    v16h b[4];
#pragma unroll
    for (int nj = 0; nj < 4; ++nj)
      b[nj] = rhn_load_frag_b(W + (size_t)(n0 + 16 * nj) * (size_t)K, K, k, half, m16);
#pragma unroll
    for (int nj = 0; nj < 4; ++nj) {
      c[0][nj] = RHN_WMMA(a0, b[nj], c[0][nj]);
      c[1][nj] = RHN_WMMA(a1, b[nj], c[1][nj]);
    }
  }

#pragma unroll
  for (int nj = 0; nj < 4; ++nj) {
    const int   col = n0 + 16 * nj + m16;
    const float bv  = bias[col];
#pragma unroll
    for (int mi = 0; mi < 2; ++mi)
#pragma unroll
      for (int r = 0; r < 8; ++r) {
        const int row = m0 + 16 * mi + r + 8 * half;
        out[(size_t)row * (size_t)N + (size_t)col] = c[mi][nj][r] + bv;
      }
  }
}

// ---- f32 -> f16 conversion (vectorized, grid-stride) ----------------------
__global__ void hyperrhn_cvt_f16(const float* __restrict__ in,
                                 _Float16* __restrict__ out, int n4) {
  int i = blockIdx.x * blockDim.x + threadIdx.x;
  const int stride = gridDim.x * blockDim.x;
  for (; i < n4; i += stride) {
    const float4 v = ((const float4*)in)[i];
    v4h h;
    h[0] = (_Float16)v.x; h[1] = (_Float16)v.y;
    h[2] = (_Float16)v.z; h[3] = (_Float16)v.w;
    ((v4h*)out)[i] = h;
  }
}

// ---------------------------------------------------------------------------
extern "C" void kernel_launch(void* const* d_in, const int* in_sizes, int n_in,
                              void* d_out, int out_size, void* d_ws, size_t ws_size,
                              hipStream_t stream) {
  (void)in_sizes; (void)n_in; (void)out_size; (void)ws_size;

  const float* x        = (const float*)d_in[0];
  const float* sN0      = (const float*)d_in[1];
  const float* sH0      = (const float*)d_in[2];
  const float* Wh_hyper = (const float*)d_in[3];
  const float* bh_hyper = (const float*)d_in[4];
  const float* Wi_hyper = (const float*)d_in[5];
  const float* bi_hyper = (const float*)d_in[6];
  const float* Wh       = (const float*)d_in[7];
  const float* bh       = (const float*)d_in[8];
  const float* Wi       = (const float*)d_in[9];
  const float* bi       = (const float*)d_in[10];
  const float* Wup      = (const float*)d_in[11];
  const float* bup      = (const float*)d_in[12];
  // d_in[13] = trainable (ignored; eval mode => dropout is identity)

  // ---- workspace layout (all chunks are multiples of 256B; base is aligned)
  char*  ws  = (char*)d_ws;
  size_t off = 0;
  auto alloc16 = [&](size_t elems) { _Float16* p = (_Float16*)(ws + off); off += elems * 2; return p; };
  auto alloc32 = [&](size_t elems) { float*    p = (float*)(ws + off);    off += elems * 4; return p; };

  _Float16* Whh16   = alloc16((size_t)RHN_DEPTH * 2 * RHN_HH * RHN_HH);
  _Float16* Wih16   = alloc16((size_t)2 * RHN_HH * RHN_ED);
  _Float16* Wh16    = alloc16((size_t)RHN_DEPTH * 2 * RHN_HN * RHN_HN);
  _Float16* Wi16    = alloc16((size_t)2 * RHN_HN * RHN_ED);
  _Float16* Wup16   = alloc16((size_t)RHN_DEPTH * RHN_HN * RHN_HH);
  _Float16* x16     = alloc16((size_t)RHN_B * RHN_ED);
  _Float16* sH16[2] = { alloc16((size_t)RHN_B * RHN_HH), alloc16((size_t)RHN_B * RHN_HH) };
  _Float16* sN16[2] = { alloc16((size_t)RHN_B * RHN_HN), alloc16((size_t)RHN_B * RHN_HN) };
  float*    sH32    = alloc32((size_t)RHN_B * RHN_HH);
  float*    sN32    = alloc32((size_t)RHN_B * RHN_HN);
  float*    z32     = alloc32((size_t)RHN_B * RHN_HN);

  auto cvt = [&](const float* in, _Float16* out, size_t n) {
    const int n4 = (int)(n / 4);
    int blocks = (n4 + 255) / 256;
    if (blocks > 2048) blocks = 2048;
    hyperrhn_cvt_f16<<<blocks, 256, 0, stream>>>(in, out, n4);
  };

  // One-time f32 -> f16 of weights + initial activations (HBM-bound pass)
  cvt(x,        x16,     (size_t)RHN_B * RHN_ED);
  cvt(sH0,      sH16[0], (size_t)RHN_B * RHN_HH);
  cvt(sN0,      sN16[0], (size_t)RHN_B * RHN_HN);
  cvt(Wh_hyper, Whh16,   (size_t)RHN_DEPTH * 2 * RHN_HH * RHN_HH);
  cvt(Wi_hyper, Wih16,   (size_t)2 * RHN_HH * RHN_ED);
  cvt(Wh,       Wh16,    (size_t)RHN_DEPTH * 2 * RHN_HN * RHN_HN);
  cvt(Wi,       Wi16,    (size_t)2 * RHN_HN * RHN_ED);
  cvt(Wup,      Wup16,   (size_t)RHN_DEPTH * RHN_HN * RHN_HH);

  for (int i = 0; i < RHN_DEPTH; ++i) {
    const int rd = i & 1, wr = (i + 1) & 1;

    // ---- hypernetwork highway step: sHyper <- gate(sHyper@Whh.T [+ x@Wih.T])
    hyperrhn_highway_wmma<<<dim3(RHN_B / 32, RHN_HH / 32), 32, 0, stream>>>(
        sH16[rd], Whh16 + (size_t)i * 2 * RHN_HH * RHN_HH,
        bh_hyper + (size_t)i * 2 * RHN_HH, RHN_HH,
        (i == 0) ? x16 : nullptr, Wih16,
        (i == 0) ? bi_hyper : nullptr, RHN_ED,
        /*zscale=*/nullptr,
        (i == 0) ? sH0 : sH32,
        sH32, sH16[wr], RHN_HH);

    // ---- hyper embedding upscale: z = sHyper @ Wup.T + bup
    hyperrhn_gemm_bias_wmma<<<dim3(RHN_B / 32, RHN_HN / 64), 32, 0, stream>>>(
        sH16[wr], Wup16 + (size_t)i * RHN_HN * RHN_HH,
        bup + (size_t)i * RHN_HN, z32, RHN_HN, RHN_HH);

    // ---- main-network highway step, z-scaled (HyperLinear)
    hyperrhn_highway_wmma<<<dim3(RHN_B / 32, RHN_HN / 32), 32, 0, stream>>>(
        sN16[rd], Wh16 + (size_t)i * 2 * RHN_HN * RHN_HN,
        bh + (size_t)i * 2 * RHN_HN, RHN_HN,
        (i == 0) ? x16 : nullptr, Wi16,
        (i == 0) ? bi : nullptr, RHN_ED,
        /*zscale=*/z32,
        (i == 0) ? sN0 : sN32,
        sN32, sN16[wr], RHN_HN);
  }

  // Outputs: (sHyper, sNetwork) concatenated flat, f32
  hipMemcpyAsync(d_out, sH32, (size_t)RHN_B * RHN_HH * sizeof(float),
                 hipMemcpyDeviceToDevice, stream);
  hipMemcpyAsync((float*)d_out + (size_t)RHN_B * RHN_HH, sN32,
                 (size_t)RHN_B * RHN_HN * sizeof(float),
                 hipMemcpyDeviceToDevice, stream);
}
